// SwiGLUMoELayer_33337536152174
// MI455X (gfx1250) — matrix operand order repaired and checked by reading, not compiled, not run on hardware
//
#include <hip/hip_runtime.h>
#include <hip/hip_bf16.h>
#include <stdint.h>

// SwiGLU MoE: router -> top2 -> grouped bf16-WMMA GEMMs (gate/up fused + down) -> combine.
// v_wmma_f32_16x16x32_bf16 (wave32, f32 acc), double-buffered LDS (1 barrier/K-step),
// paired v_cvt_pk_bf16_f32 conversion, b128 LDS fragment traffic.

#define D_MODEL 1024
#define D_FF    4096
#define N_EXP   8
#define TOPK    2
#define BATCH   4096
#define NPAIR   (BATCH * TOPK)

#define BM 64
#define BN 64
#define BK 32
#define LSTR 40   // LDS row stride in bf16 elems: 80B rows -> every row start 16B-aligned

typedef __bf16        v16bf  __attribute__((ext_vector_type(16)));
typedef float         v8f    __attribute__((ext_vector_type(8)));
typedef unsigned int  u32x4  __attribute__((ext_vector_type(4)));
typedef __bf16        bf16x2 __attribute__((ext_vector_type(2)));
typedef float         f32x2  __attribute__((ext_vector_type(2)));

union Frag { u32x4 q[2]; v16bf v; };

__device__ __forceinline__ unsigned short f2bf(float f) {
  union { __bf16 b; unsigned short s; } c; c.b = (__bf16)f; return c.s;
}
// one v_cvt_pk_bf16_f32: packs {lo, hi} into a dword
__device__ __forceinline__ unsigned int pk2(float lo, float hi) {
  f32x2 f = {lo, hi};
  union { bf16x2 b; unsigned int u; } c;
  c.b = __builtin_convertvector(f, bf16x2);
  return c.u;
}

// ---------------- Router: one wave (32 lanes) per token ----------------
__global__ __launch_bounds__(256)
void moe_router(const float* __restrict__ x, const float* __restrict__ rw,
                const float* __restrict__ rb,
                int* __restrict__ topk_idx, float* __restrict__ topk_w) {
  int tid  = threadIdx.x;
  int lane = tid & 31;
  int t    = blockIdx.x * 8 + (tid >> 5);

  float xv[32];
  #pragma unroll
  for (int j = 0; j < 32; ++j) xv[j] = x[(long)t * D_MODEL + j * 32 + lane];

  float logit[N_EXP];
  #pragma unroll
  for (int e = 0; e < N_EXP; ++e) {
    float acc = 0.f;
    #pragma unroll
    for (int j = 0; j < 32; ++j) acc += xv[j] * rw[e * D_MODEL + j * 32 + lane];
    #pragma unroll
    for (int off = 16; off > 0; off >>= 1) acc += __shfl_xor(acc, off, 32);
    logit[e] = acc + rb[e];
  }
  int i0 = 0, i1 = 1;
  float v0 = logit[0], v1 = -3.4e38f;
  #pragma unroll
  for (int e = 1; e < N_EXP; ++e) {
    float l = logit[e];
    if (l > v0)      { v1 = v0; i1 = i0; v0 = l; i0 = e; }
    else if (l > v1) { v1 = l;  i1 = e; }
  }
  float w0 = 1.f / (1.f + __expf(v1 - v0));   // renormalized softmax top-2; Z cancels
  if (lane == 0) {
    topk_idx[t * 2 + 0] = i0; topk_idx[t * 2 + 1] = i1;
    topk_w[t * 2 + 0]   = w0; topk_w[t * 2 + 1]   = 1.f - w0;
  }
}

// ---------------- Expert bucketing ----------------
__global__ void moe_count(const int* __restrict__ topk_idx, int* __restrict__ counts) {
  int i = blockIdx.x * blockDim.x + threadIdx.x;
  if (i < NPAIR) atomicAdd(&counts[topk_idx[i]], 1);
}
__global__ void moe_offsets(const int* __restrict__ counts, int* __restrict__ offs) {
  if (threadIdx.x == 0) {
    int s = 0;
    for (int e = 0; e < N_EXP; ++e) { offs[e] = s; s += counts[e]; }
    offs[N_EXP] = s;
  }
}
__global__ void moe_scatter(const int* __restrict__ topk_idx, const int* __restrict__ offs,
                            int* __restrict__ cursor, int* __restrict__ slot_pair) {
  int i = blockIdx.x * blockDim.x + threadIdx.x;
  if (i < NPAIR) {
    int e = topk_idx[i];
    int pos = atomicAdd(&cursor[e], 1);
    slot_pair[offs[e] + pos] = i;   // pair id = token*2 + k
  }
}

// ---------------- Grouped GEMM 1: gate+up fused, SwiGLU epilogue ----------------
__global__ __launch_bounds__(256)
void moe_gateup(const float* __restrict__ x, const float* __restrict__ wg,
                const float* __restrict__ wu, const int* __restrict__ offs,
                const int* __restrict__ slot_pair, unsigned short* __restrict__ hidden) {
  int e    = blockIdx.z;
  int rows = offs[e + 1] - offs[e];
  int m0   = blockIdx.y * BM;
  if (m0 >= rows) return;                 // early-exit padding blocks
  int n0        = blockIdx.x * BN;
  int base_slot = offs[e] + m0;

  __shared__ __align__(16) unsigned short As[2][BM * LSTR];   // [m][k] bf16
  __shared__ __align__(16) unsigned short Bg[2][BN * LSTR];   // [n][k] bf16 (K-transposed)
  __shared__ __align__(16) unsigned short Bu[2][BN * LSTR];

  int tid  = threadIdx.x;
  int lane = tid & 31;
  int wave = tid >> 5;
  int mt   = wave & 3;                    // wave's 16-row tile within 64
  int ntb  = (wave >> 2) * 32;            // wave covers n tiles {ntb, ntb+16}

  // A loader: 4 threads per row, 8 consecutive k each (2x b128 global)
  int arow = tid >> 2, akc = (tid & 3) * 8;
  // B loader: 2 k-rows x 4 n-cols per thread -> (k,k+1) dword pairs into LDS
  int bk2 = (tid & 15) * 2, bnc4 = (tid >> 4) * 4;

  int tok = 0; unsigned int amask = 0u;
  if (m0 + arow < rows) { tok = slot_pair[base_slot + arow] >> 1; amask = 0xffffffffu; }
  const float* xrow = x + (long)tok * D_MODEL + akc;

  v8f accG0 = {}, accG1 = {}, accU0 = {}, accU1 = {};

  float4 xa0, xa1, g0, g1, u0, u1;
  auto stage = [&](int k0) {
    xa0 = *(const float4*)(xrow + k0);
    xa1 = *(const float4*)(xrow + k0 + 4);
    long gb = ((long)e * D_MODEL + k0 + bk2) * D_FF + n0 + bnc4;
    g0 = *(const float4*)(wg + gb); g1 = *(const float4*)(wg + gb + D_FF);
    u0 = *(const float4*)(wu + gb); u1 = *(const float4*)(wu + gb + D_FF);
  };
  auto cvtstore = [&](int buf) {
    u32x4 apk;
    apk.x = pk2(xa0.x, xa0.y) & amask; apk.y = pk2(xa0.z, xa0.w) & amask;
    apk.z = pk2(xa1.x, xa1.y) & amask; apk.w = pk2(xa1.z, xa1.w) & amask;
    *(u32x4*)&As[buf][arow * LSTR + akc] = apk;
    float g0v[4], g1v[4], u0v[4], u1v[4];
    *(float4*)g0v = g0; *(float4*)g1v = g1;
    *(float4*)u0v = u0; *(float4*)u1v = u1;
    #pragma unroll
    for (int j = 0; j < 4; ++j) {
      *(unsigned int*)&Bg[buf][(bnc4 + j) * LSTR + bk2] = pk2(g0v[j], g1v[j]);
      *(unsigned int*)&Bu[buf][(bnc4 + j) * LSTR + bk2] = pk2(u0v[j], u1v[j]);
    }
  };

  const int nIter = D_MODEL / BK;
  stage(0);
  cvtstore(0);
  for (int it = 0; it < nIter; ++it) {
    int buf = it & 1;
    if (it + 1 < nIter) stage((it + 1) * BK);   // globals fly over compute
    __syncthreads();                            // tile `it` visible; buf^1 free to write

    int h = (lane < 16) ? 0 : 1;
    Frag a, bg0, bg1, bu0, bu1;
    {   // A fragment (contiguous k = 16h + s, same order as B): two b128 loads
      const u32x4* ap = (const u32x4*)&As[buf][(mt * 16 + (lane & 15)) * LSTR];
      a.q[0] = ap[2 * h]; a.q[1] = ap[2 * h + 1];
    }
    {   // B fragments (lanes<16: K=0..15 -> q0,q1; lanes>=16: K=16..31 -> q2,q3)
      const u32x4* bp0 = (const u32x4*)&Bg[buf][(ntb + (lane & 15)) * LSTR];
      const u32x4* bp1 = (const u32x4*)&Bg[buf][(ntb + 16 + (lane & 15)) * LSTR];
      const u32x4* up0 = (const u32x4*)&Bu[buf][(ntb + (lane & 15)) * LSTR];
      const u32x4* up1 = (const u32x4*)&Bu[buf][(ntb + 16 + (lane & 15)) * LSTR];
      bg0.q[0] = bp0[2 * h]; bg0.q[1] = bp0[2 * h + 1];
      bg1.q[0] = bp1[2 * h]; bg1.q[1] = bp1[2 * h + 1];
      bu0.q[0] = up0[2 * h]; bu0.q[1] = up0[2 * h + 1];
      bu1.q[0] = up1[2 * h]; bu1.q[1] = up1[2 * h + 1];
    }
    accG0 = __builtin_amdgcn_wmma_f32_16x16x32_bf16(false, a.v, false, bg0.v, (short)0, accG0, false, false);
    accG1 = __builtin_amdgcn_wmma_f32_16x16x32_bf16(false, a.v, false, bg1.v, (short)0, accG1, false, false);
    accU0 = __builtin_amdgcn_wmma_f32_16x16x32_bf16(false, a.v, false, bu0.v, (short)0, accU0, false, false);
    accU1 = __builtin_amdgcn_wmma_f32_16x16x32_bf16(false, a.v, false, bu1.v, (short)0, accU1, false, false);

    if (it + 1 < nIter) cvtstore(buf ^ 1);
  }

  // SwiGLU epilogue; C/D layout: VGPR r -> M = r (+8 upper half-wave), N = lane&15
  int mhi  = (lane >= 16) ? 8 : 0;
  int ncol = lane & 15;
  #pragma unroll
  for (int r = 0; r < 8; ++r) {
    int m_loc = mt * 16 + r + mhi;
    if (m0 + m_loc < rows) {
      long hrow = (long)(base_slot + m_loc) * D_FF + n0;
      float gA = accG0[r], uA = accU0[r];
      float gB = accG1[r], uB = accU1[r];
      hidden[hrow + ntb + ncol]      = f2bf((gA / (1.f + __expf(-gA))) * uA);
      hidden[hrow + ntb + 16 + ncol] = f2bf((gB / (1.f + __expf(-gB))) * uB);
    }
  }
}

// ---------------- Grouped GEMM 2: down projection, weighted per-pair store ----------------
__global__ __launch_bounds__(256)
void moe_down(const unsigned short* __restrict__ hidden, const float* __restrict__ wd,
              const int* __restrict__ offs, const int* __restrict__ slot_pair,
              const float* __restrict__ topk_w, float* __restrict__ down_buf) {
  int e    = blockIdx.z;
  int rows = offs[e + 1] - offs[e];
  int m0   = blockIdx.y * BM;
  if (m0 >= rows) return;
  int n0        = blockIdx.x * BN;
  int base_slot = offs[e] + m0;

  __shared__ __align__(16) unsigned short As[2][BM * LSTR];
  __shared__ __align__(16) unsigned short Bs[2][BN * LSTR];

  int tid = threadIdx.x, lane = tid & 31, wave = tid >> 5;
  int mt = wave & 3, ntb = (wave >> 2) * 32;

  int arow = tid >> 2, akc = (tid & 3) * 8;
  int bk2 = (tid & 15) * 2, bnc4 = (tid >> 4) * 4;
  unsigned int amask = (m0 + arow < rows) ? 0xffffffffu : 0u;
  int srow = base_slot + arow; if (srow > NPAIR - 1) srow = NPAIR - 1;   // clamp padded rows
  const unsigned short* hrow = hidden + (long)srow * D_FF + akc;

  v8f acc0 = {}, acc1 = {};

  u32x4 aq; float4 b0, b1;
  auto stage = [&](int k0) {
    aq = *(const u32x4*)(hrow + k0);
    long gb = ((long)e * D_FF + k0 + bk2) * D_MODEL + n0 + bnc4;
    b0 = *(const float4*)(wd + gb); b1 = *(const float4*)(wd + gb + D_MODEL);
  };
  auto cvtstore = [&](int buf) {
    *(u32x4*)&As[buf][arow * LSTR + akc] = aq & amask;
    float b0v[4], b1v[4];
    *(float4*)b0v = b0; *(float4*)b1v = b1;
    #pragma unroll
    for (int j = 0; j < 4; ++j)
      *(unsigned int*)&Bs[buf][(bnc4 + j) * LSTR + bk2] = pk2(b0v[j], b1v[j]);
  };

  const int nIter = D_FF / BK;
  stage(0);
  cvtstore(0);
  for (int it = 0; it < nIter; ++it) {
    int buf = it & 1;
    if (it + 1 < nIter) stage((it + 1) * BK);
    __syncthreads();

    int h = (lane < 16) ? 0 : 1;
    Frag a, f0, f1;
    {
      const u32x4* ap = (const u32x4*)&As[buf][(mt * 16 + (lane & 15)) * LSTR];
      a.q[0] = ap[2 * h]; a.q[1] = ap[2 * h + 1];
    }
    {
      const u32x4* bp0 = (const u32x4*)&Bs[buf][(ntb + (lane & 15)) * LSTR];
      const u32x4* bp1 = (const u32x4*)&Bs[buf][(ntb + 16 + (lane & 15)) * LSTR];
      f0.q[0] = bp0[2 * h]; f0.q[1] = bp0[2 * h + 1];
      f1.q[0] = bp1[2 * h]; f1.q[1] = bp1[2 * h + 1];
    }
    acc0 = __builtin_amdgcn_wmma_f32_16x16x32_bf16(false, a.v, false, f0.v, (short)0, acc0, false, false);
    acc1 = __builtin_amdgcn_wmma_f32_16x16x32_bf16(false, a.v, false, f1.v, (short)0, acc1, false, false);

    if (it + 1 < nIter) cvtstore(buf ^ 1);
  }

  int mhi = (lane >= 16) ? 8 : 0;
  int ncol = lane & 15;
  #pragma unroll
  for (int r = 0; r < 8; ++r) {
    int m_loc = mt * 16 + r + mhi;
    if (m0 + m_loc < rows) {
      int pair = slot_pair[base_slot + m_loc];
      float w  = topk_w[pair];
      long orow = (long)pair * D_MODEL + n0;
      down_buf[orow + ntb + ncol]      = acc0[r] * w;
      down_buf[orow + ntb + 16 + ncol] = acc1[r] * w;
    }
  }
}

// ---------------- Combine the two expert contributions per token ----------------
__global__ void moe_combine(const float* __restrict__ down_buf, float* __restrict__ out) {
  int i = blockIdx.x * blockDim.x + threadIdx.x;   // over BATCH*D_MODEL/4
  int t = i >> 8, d4 = i & 255;                    // 256 float4 per token row
  const float4* db = (const float4*)down_buf;
  float4 a = db[(long)t * 512 + d4];
  float4 b = db[(long)t * 512 + 256 + d4];
  float4 r; r.x = a.x + b.x; r.y = a.y + b.y; r.z = a.z + b.z; r.w = a.w + b.w;
  ((float4*)out)[i] = r;
}

extern "C" void kernel_launch(void* const* d_in, const int* in_sizes, int n_in,
                              void* d_out, int out_size, void* d_ws, size_t ws_size,
                              hipStream_t stream) {
  (void)in_sizes; (void)n_in; (void)out_size; (void)ws_size;
  const float* x  = (const float*)d_in[0];
  const float* rw = (const float*)d_in[1];
  const float* rb = (const float*)d_in[2];
  const float* wg = (const float*)d_in[3];
  const float* wu = (const float*)d_in[4];
  const float* wd = (const float*)d_in[5];
  float* out = (float*)d_out;

  char* p = (char*)d_ws;
  int*   topk_idx  = (int*)p;   p += NPAIR * sizeof(int);
  float* topk_w    = (float*)p; p += NPAIR * sizeof(float);
  int*   counts    = (int*)p;   p += N_EXP * sizeof(int);
  int*   cursor    = (int*)p;   p += N_EXP * sizeof(int);
  int*   offs      = (int*)p;   p += 16 * sizeof(int);
  int*   slot_pair = (int*)p;   p += NPAIR * sizeof(int);
  p = (char*)(((uintptr_t)p + 255) & ~(uintptr_t)255);
  unsigned short* hidden = (unsigned short*)p; p += (size_t)NPAIR * D_FF * sizeof(unsigned short);
  float* down_buf = (float*)p;

  hipMemsetAsync(counts, 0, 2 * N_EXP * sizeof(int), stream);   // counts + cursor
  moe_router <<<BATCH / 8,  256, 0, stream>>>(x, rw, rb, topk_idx, topk_w);
  moe_count  <<<NPAIR / 256, 256, 0, stream>>>(topk_idx, counts);
  moe_offsets<<<1, 32, 0, stream>>>(counts, offs);
  moe_scatter<<<NPAIR / 256, 256, 0, stream>>>(topk_idx, offs, cursor, slot_pair);
  moe_gateup <<<dim3(D_FF / BN,    NPAIR / BM, N_EXP), 256, 0, stream>>>(x, wg, wu, offs, slot_pair, hidden);
  moe_down   <<<dim3(D_MODEL / BN, NPAIR / BM, N_EXP), 256, 0, stream>>>(hidden, wd, offs, slot_pair, topk_w, down_buf);
  moe_combine<<<(BATCH * D_MODEL) / (256 * 4), 256, 0, stream>>>(down_buf, out);
}
